// GethConsensus_51470888075730
// MI455X (gfx1250) — matrix operand order repaired and checked by reading, not compiled, not run on hardware
//
#include <hip/hip_runtime.h>

// ---------------------------------------------------------------------------
// MI455X (gfx1250, wave32) sparse-MLP forward:
//   hid = relu(x @ W_alive^T scattered into [128,2048] + bias); out = hid @ fc2_w^T + b
// All matrix math on v_wmma_f32_16x16x32_f16 (f32 accumulate), f32->f16
// conversion done in VGPRs, operand loads are contiguous b128 per lane per the
// CDNA5 WMMA VGPR layouts (CDNA5 ISA 7.12.2).
// ---------------------------------------------------------------------------

typedef __attribute__((ext_vector_type(16))) _Float16 v16h;
typedef __attribute__((ext_vector_type(8)))  _Float16 v8h;
typedef __attribute__((ext_vector_type(8)))  float    v8f;
typedef __attribute__((ext_vector_type(4)))  float    v4f;

#define IN_SIZE 784
#define HID     2048
#define N_ALIVE 1536
#define BATCH   128
#define NCLS    10

// -------------------------------------------------------------------------
// Kernel 1: hid[b,h] = relu(bias[h]) for every (b,h); GEMM overwrites alive h.
// -------------------------------------------------------------------------
__global__ __launch_bounds__(256) void gc_init_hid(
    const float* __restrict__ bias, _Float16* __restrict__ hidh) {
  int idx = blockIdx.x * 256 + threadIdx.x;       // 128*2048 threads
  int h = idx & (HID - 1);
  hidh[idx] = (_Float16)fmaxf(bias[h], 0.0f);
}

// -------------------------------------------------------------------------
// Kernel 2: hid[:, alive] = relu(x @ W^T + bias[alive]).
// One wave per 16x16 C tile: 8 M-tiles x 96 N-tiles = 768 tiles.
// K = 784 = 24 * 32 + 16 (tail zero-padded in-register, exact).
// -------------------------------------------------------------------------
__global__ __launch_bounds__(256) void gc_spmm_wmma(
    const float* __restrict__ x,      // [128,784]
    const float* __restrict__ w,      // [1536,784]  (the COO "values", dense)
    const float* __restrict__ bias,   // [2048]
    const int*   __restrict__ rows,   // rows[a*784] = alive hidden index of dense row a
    _Float16*    __restrict__ hidh)   // [128,2048] f16 scratch
{
  const int wave = threadIdx.x >> 5;
  const int lane = threadIdx.x & 31;
  const int lrow = lane & 15;
  const int lhi  = lane >> 4;                 // 0: lanes 0-15, 1: lanes 16-31

  const int tile = blockIdx.x * 8 + wave;     // 0..767
  const int tm   = tile & 7;                  // M tile (batch)
  const int tn   = tile >> 3;                 // N tile (alive rows)

  const float* xr = x + (tm * 16 + lrow) * IN_SIZE;   // A: row M = lrow
  const int    n  = tn * 16 + lrow;
  const float* wr = w + n * IN_SIZE;                  // B: col N = lrow (W row = B^T)
  const int aoff = lhi * 8;                           // A half-lane K offset
  const int koff = lhi * 16;                          // unused (B loads use aoff, like A)

  v8f c = {};
  for (int kc = 0; kc < 25; ++kc) {
    const int  k    = kc * 32;
    const bool tail = (kc == 24);            // K chunk 768..799, valid K < 784

    // A fragment: halves 0..7 = K k+aoff.., halves 8..15 = K k+16+aoff..
    v4f a0 = *(const v4f*)(xr + k + aoff);
    v4f a1 = *(const v4f*)(xr + k + aoff + 4);
    v4f a2 = {}, a3 = {};
    if (!tail) {                              // K >= 784 -> zero (uniform branch)
      a2 = *(const v4f*)(xr + k + 16 + aoff);
      a3 = *(const v4f*)(xr + k + 16 + aoff + 4);
    }
    // B fragment: halves 0..7 = K k+aoff.., halves 8..15 = K k+16+aoff..
    v4f b0 = *(const v4f*)(wr + k + aoff);
    v4f b1 = *(const v4f*)(wr + k + aoff + 4);
    v4f b2 = {}, b3 = {};
    if (!tail) {                              // K >= 784 -> zero (uniform branch)
      b2 = *(const v4f*)(wr + k + 16 + aoff);
      b3 = *(const v4f*)(wr + k + 16 + aoff + 4);
    }

    v16h af, bf;
#pragma unroll
    for (int i = 0; i < 4; ++i) {
      af[i]      = (_Float16)a0[i];
      af[4 + i]  = (_Float16)a1[i];
      af[8 + i]  = (_Float16)a2[i];
      af[12 + i] = (_Float16)a3[i];
      bf[i]      = (_Float16)b0[i];
      bf[4 + i]  = (_Float16)b1[i];
      bf[8 + i]  = (_Float16)b2[i];
      bf[12 + i] = (_Float16)b3[i];
    }
    c = __builtin_amdgcn_wmma_f32_16x16x32_f16(
        false, af, false, bf, (short)0, c, false, false);
  }

  // Scatter epilogue: C VGPR r -> M = tm*16 + r + lhi*8, N = lrow -> h = alive[n]
  const int   h  = rows[n * IN_SIZE];
  const float bv = bias[h];
#pragma unroll
  for (int r = 0; r < 8; ++r) {
    const int m = tm * 16 + r + lhi * 8;
    hidh[m * HID + h] = (_Float16)fmaxf(c[r] + bv, 0.0f);
  }
}

// -------------------------------------------------------------------------
// Kernel 3: out[128,10] = hid @ fc2_w^T + fc2_b.  N padded 10->16 with zero
// B lanes. One block, 8 waves = 8 M-tiles, K = 2048 = 64 chunks of 32.
// -------------------------------------------------------------------------
__global__ __launch_bounds__(256) void gc_head_wmma(
    const _Float16* __restrict__ hidh,  // [128,2048] f16
    const float*    __restrict__ w2,    // [10,2048]
    const float*    __restrict__ b2,    // [10]
    float*          __restrict__ out)   // [128,10]
{
  const int wave = threadIdx.x >> 5;    // M tile
  const int lane = threadIdx.x & 31;
  const int lrow = lane & 15;
  const int lhi  = lane >> 4;

  const _Float16* ar = hidh + (wave * 16 + lrow) * HID;
  const int  aoff   = lhi * 8;
  const int  koff   = lhi * 16;
  const bool bvalid = (lrow < NCLS);
  const float* wrp  = w2 + lrow * HID;

  v8f c = {};
  for (int kc = 0; kc < 64; ++kc) {
    const int k = kc * 32;
    v8h alo = *(const v8h*)(ar + k + koff);
    v8h ahi = *(const v8h*)(ar + k + koff + 8);
    v16h af, bf = {};
#pragma unroll
    for (int i = 0; i < 8; ++i) { af[i] = alo[i]; af[8 + i] = ahi[i]; }
    if (bvalid) {
      v4f b0 = *(const v4f*)(wrp + k + koff);
      v4f b1 = *(const v4f*)(wrp + k + koff + 4);
      v4f b2v = *(const v4f*)(wrp + k + koff + 8);
      v4f b3 = *(const v4f*)(wrp + k + koff + 12);
#pragma unroll
      for (int i = 0; i < 4; ++i) {
        bf[i]      = (_Float16)b0[i];
        bf[4 + i]  = (_Float16)b1[i];
        bf[8 + i]  = (_Float16)b2v[i];
        bf[12 + i] = (_Float16)b3[i];
      }
    }
    c = __builtin_amdgcn_wmma_f32_16x16x32_f16(
        false, af, false, bf, (short)0, c, false, false);
  }

  if (bvalid) {
    const float bb = b2[lrow];
#pragma unroll
    for (int r = 0; r < 8; ++r) {
      const int m = wave * 16 + r + lhi * 8;
      out[m * NCLS + lrow] = c[r] + bb;
    }
  }
}

// -------------------------------------------------------------------------
extern "C" void kernel_launch(void* const* d_in, const int* in_sizes, int n_in,
                              void* d_out, int out_size, void* d_ws, size_t ws_size,
                              hipStream_t stream) {
  const float* x     = (const float*)d_in[0];   // [128,784]
  const float* vals  = (const float*)d_in[1];   // [1536*784]
  const float* sbias = (const float*)d_in[2];   // [2048]
  const float* fc2w  = (const float*)d_in[3];   // [10,2048]
  const float* fc2b  = (const float*)d_in[4];   // [10]
  const int*   rows  = (const int*)d_in[5];     // [nnz]
  float*       out   = (float*)d_out;           // [128,10]

  _Float16* hidh = (_Float16*)d_ws;             // 128*2048*2 = 512 KB scratch

  gc_init_hid<<<(BATCH * HID) / 256, 256, 0, stream>>>(sbias, hidh);
  gc_spmm_wmma<<<96, 256, 0, stream>>>(x, vals, sbias, rows, hidh);
  gc_head_wmma<<<1, 256, 0, stream>>>(hidh, fc2w, fc2b, out);
}
